// QuantumFeedForward_5927054868481
// MI455X (gfx1250) — compile-verified, matrix operand order checked by reading, not run on hardware
//
#include <hip/hip_runtime.h>

typedef __attribute__((ext_vector_type(16))) _Float16 v16h;
typedef __attribute__((ext_vector_type(8)))  _Float16 v8h;
typedef __attribute__((ext_vector_type(8)))  float    v8f;

#define D_MODEL   512
#define DIM       1024
#define N_QUBITS  10
#define N_LAYERS  2
#define N_TOKENS  16384
#define TOK_PER_BLK 16
#define THREADS   256

#define NT1 (DIM / 16)      /* 64 N-tiles, GEMM1 */
#define KT1 (D_MODEL / 32)  /* 16 k-steps        */
#define NT2 (D_MODEL / 16)  /* 32 N-tiles, GEMM2 */
#define KT2 (DIM / 32)      /* 32 k-steps        */

// K position of fragment element e for a given lane, per CDNA5 ISA 7.12.2:
// lanes 0-15 hold K {0..7, 16..23}; lanes 16-31 hold K {8..15, 24..31}.
__device__ __forceinline__ int kpat(int e, int lane) {
  return (e & 7) + ((e >> 3) << 4) + ((lane >> 4) << 3);
}

// Repack W_in [1024,512] (row n, col k) into WMMA-B fragment order (f16).
// Linear fragment index = ((nt*KT1 + kt)*32 + lane)*16 + e.
__global__ void prep_win(const float* __restrict__ W_in, _Float16* __restrict__ bf) {
  int gid = blockIdx.x * blockDim.x + threadIdx.x;
  int e = gid & 15, lane = (gid >> 4) & 31, kt = (gid >> 9) & 15, nt = gid >> 13;
  int n = nt * 16 + (lane & 15);
  int k = kt * 32 + kpat(e, lane);
  bf[gid] = (_Float16)W_in[(size_t)n * D_MODEL + k];
}

// Repack W_out [512,1024] (row d=n, col k) into WMMA-B fragment order (f16).
__global__ void prep_wout(const float* __restrict__ W_out, _Float16* __restrict__ bf) {
  int gid = blockIdx.x * blockDim.x + threadIdx.x;
  int e = gid & 15, lane = (gid >> 4) & 31, kt = (gid >> 9) & 31, nt = gid >> 14;
  int n = nt * 16 + (lane & 15);
  int k = kt * 32 + kpat(e, lane);
  bf[gid] = (_Float16)W_out[(size_t)n * DIM + k];
}

struct cpx { float x, y; };
__device__ __forceinline__ cpx cmul(cpx a, cpx b) {
  return { a.x * b.x - a.y * b.y, a.x * b.y + a.y * b.x };
}
__device__ __forceinline__ cpx cadd(cpx a, cpx b) { return { a.x + b.x, a.y + b.y }; }

// LDS layout (bytes):
//   [0      , 65536 )  psi_re  16 x 1024 f32
//   [65536  , 131072)  psi_im  16 x 1024 f32
//   [131072 , 163840)  stage_h : x-tile f16 (16KB) then probs f16 (32KB)
//   [163840 , 196608)  out_t   16 x 512 f32
//   [196608 , 197632)  red1    256 f32
//   [197632 , 198656)  red2    256 f32
//   [198656 , 198720)  stat_a  16 f32
//   [198720 , 198784)  stat_b  16 f32
#define LDS_BYTES 198784

__global__ __launch_bounds__(THREADS)
void qff_main(const float* __restrict__ x,
              const _Float16* __restrict__ bf1,
              const _Float16* __restrict__ bf2,
              const float* __restrict__ b_in,
              const float* __restrict__ b_out,
              const float* __restrict__ rot,
              const float* __restrict__ ent,
              const float* __restrict__ ln_w,
              const float* __restrict__ ln_b,
              float* __restrict__ out)
{
  extern __shared__ __align__(16) char smem[];
  float*    psi_re  = (float*)smem;
  float*    psi_im  = (float*)(smem + 65536);
  _Float16* stage_h = (_Float16*)(smem + 131072);
  float*    out_t   = (float*)(smem + 163840);
  float*    red1    = (float*)(smem + 196608);
  float*    red2    = (float*)(smem + 197632);
  float*    stat_a  = (float*)(smem + 198656);
  float*    stat_b  = (float*)(smem + 198720);

  const int tid  = threadIdx.x;
  const int lane = tid & 31;
  const int w    = tid >> 5;       // wave id, 0..7
  const int hi   = lane >> 4;      // half-wave select
  const int mrow = lane & 15;      // A-row / B-col within tile
  const int t0   = blockIdx.x * TOK_PER_BLK;

  // ---------- Phase 1a: stage x tile -> LDS f16 [16][512] ----------
  for (int i = tid; i < TOK_PER_BLK * D_MODEL; i += THREADS) {
    int m = i >> 9, k = i & (D_MODEL - 1);
    stage_h[i] = (_Float16)x[(size_t)(t0 + m) * D_MODEL + k];
  }
  __syncthreads();

  // ---------- Phase 1b: xq = x @ W_in^T + b_in via WMMA ----------
  for (int ti = 0; ti < NT1 / 8; ++ti) {            // 8 N-tiles per wave
    int nt = w * (NT1 / 8) + ti;
    v8f acc = {};
    for (int kt = 0; kt < KT1; ++kt) {
      union { v16h v; v8h h[2]; } a;
      const _Float16* ap = stage_h + mrow * D_MODEL + kt * 32 + hi * 8;
      a.h[0] = *(const v8h*)(ap);        // K {0..7}  (+8 for hi half)
      a.h[1] = *(const v8h*)(ap + 16);   // K {16..23}(+8 for hi half)
      v16h b = *(const v16h*)(bf1 + ((size_t)(nt * KT1 + kt) * 32 + lane) * 16);
      acc = __builtin_amdgcn_wmma_f32_16x16x32_f16(false, a.v, false, b,
                                                   (short)0, acc, false, false);
    }
    int n = nt * 16 + mrow;
    float bias = b_in[n];
    #pragma unroll
    for (int r = 0; r < 8; ++r) {
      int m = r + hi * 8;                // C/D layout: M = reg + 8*(lane/16)
      psi_re[m * DIM + n] = acc[r] + bias;
    }
  }
  __syncthreads();

  // ---------- Phase 1c: per-token L2 normalize ----------
  {
    int m = tid >> 4, c0 = tid & 15;
    float s = 0.f;
    for (int c = c0; c < DIM; c += 16) { float v = psi_re[m * DIM + c]; s += v * v; }
    red1[tid] = s;
  }
  __syncthreads();
  if (tid < TOK_PER_BLK) {
    float s = 0.f;
    for (int k = 0; k < 16; ++k) s += red1[tid * 16 + k];
    stat_a[tid] = 1.0f / fmaxf(sqrtf(s), 1e-12f);
  }
  __syncthreads();
  for (int i = tid; i < TOK_PER_BLK * DIM; i += THREADS) {
    int m = i >> 10;
    psi_re[i] *= stat_a[m];
    psi_im[i]  = 0.f;
  }
  __syncthreads();

  // ---------- Phase 2: quantum circuit, state resident in LDS ----------
  for (int l = 0; l < N_LAYERS; ++l) {
    for (int q = 0; q < N_QUBITS; ++q) {
      const float* r3 = rot + ((l * N_QUBITS) + q) * 3;
      float h0 = 0.5f * r3[0], h1 = 0.5f * r3[1], h2 = 0.5f * r3[2];
      float c0 = cosf(h0), s0 = sinf(h0);
      float c1 = cosf(h1), s1 = sinf(h1);
      // U = Rz(t2) * Ry(t1) * Rx(t0)
      cpx rx00 = {c0, 0.f}, rx01 = {0.f, -s0}, rx10 = {0.f, -s0}, rx11 = {c0, 0.f};
      cpx m00 = cadd(cmul({ c1, 0.f}, rx00), cmul({-s1, 0.f}, rx10));
      cpx m01 = cadd(cmul({ c1, 0.f}, rx01), cmul({-s1, 0.f}, rx11));
      cpx m10 = cadd(cmul({ s1, 0.f}, rx00), cmul({ c1, 0.f}, rx10));
      cpx m11 = cadd(cmul({ s1, 0.f}, rx01), cmul({ c1, 0.f}, rx11));
      cpx rz0 = {cosf(h2), -sinf(h2)};
      cpx rz1 = {cosf(h2),  sinf(h2)};
      cpx u00 = cmul(rz0, m00), u01 = cmul(rz0, m01);
      cpx u10 = cmul(rz1, m10), u11 = cmul(rz1, m11);

      int b    = N_QUBITS - 1 - q;       // qubit 0 is MSB
      int mask = (1 << b) - 1;
      for (int p = tid; p < TOK_PER_BLK * (DIM / 2); p += THREADS) {
        int m  = p >> 9, j = p & (DIM / 2 - 1);
        int i0 = ((j & ~mask) << 1) | (j & mask);
        int i1 = i0 | (1 << b);
        int base = m * DIM;
        cpx a0 = { psi_re[base + i0], psi_im[base + i0] };
        cpx a1 = { psi_re[base + i1], psi_im[base + i1] };
        cpx n0 = cadd(cmul(u00, a0), cmul(u01, a1));
        cpx n1 = cadd(cmul(u10, a0), cmul(u11, a1));
        psi_re[base + i0] = n0.x; psi_im[base + i0] = n0.y;
        psi_re[base + i1] = n1.x; psi_im[base + i1] = n1.y;
      }
      __syncthreads();
    }
    // 9 nearest-neighbor controlled-phase gates fused into one diagonal pass
    for (int i = tid; i < TOK_PER_BLK * DIM; i += THREADS) {
      int basis = i & (DIM - 1);
      float ph = 0.f;
      #pragma unroll
      for (int t = 0; t < N_QUBITS - 1; ++t) {
        int b1 = N_QUBITS - 1 - t, b2 = b1 - 1;
        if (((basis >> b1) & (basis >> b2)) & 1) ph += ent[l * (N_QUBITS - 1) + t];
      }
      float cc = cosf(ph), ss = sinf(ph);
      float re = psi_re[i], im = psi_im[i];
      psi_re[i] = re * cc - im * ss;
      psi_im[i] = re * ss + im * cc;
    }
    __syncthreads();
  }

  // ---------- Phase 3: probs = |psi|^2 -> f16 LDS ----------
  for (int i = tid; i < TOK_PER_BLK * DIM; i += THREADS) {
    float re = psi_re[i], im = psi_im[i];
    stage_h[i] = (_Float16)(re * re + im * im);
  }
  __syncthreads();

  // ---------- Phase 4: out = probs @ W_out^T + b_out via WMMA ----------
  for (int ti = 0; ti < NT2 / 8; ++ti) {            // 4 N-tiles per wave
    int nt = w * (NT2 / 8) + ti;
    v8f acc = {};
    for (int kt = 0; kt < KT2; ++kt) {
      union { v16h v; v8h h[2]; } a;
      const _Float16* ap = stage_h + mrow * DIM + kt * 32 + hi * 8;
      a.h[0] = *(const v8h*)(ap);
      a.h[1] = *(const v8h*)(ap + 16);
      v16h b = *(const v16h*)(bf2 + ((size_t)(nt * KT2 + kt) * 32 + lane) * 16);
      acc = __builtin_amdgcn_wmma_f32_16x16x32_f16(false, a.v, false, b,
                                                   (short)0, acc, false, false);
    }
    int n = nt * 16 + mrow;
    float bias = b_out[n];
    #pragma unroll
    for (int r = 0; r < 8; ++r) {
      int m = r + hi * 8;
      out_t[m * D_MODEL + n] = acc[r] + bias;
    }
  }
  __syncthreads();

  // ---------- Phase 5: LayerNorm over d_model and write out ----------
  {
    int m = tid >> 4, c0 = tid & 15;
    float s = 0.f, ss = 0.f;
    for (int c = c0; c < D_MODEL; c += 16) {
      float v = out_t[m * D_MODEL + c];
      s += v; ss += v * v;
    }
    red1[tid] = s; red2[tid] = ss;
  }
  __syncthreads();
  if (tid < TOK_PER_BLK) {
    float s = 0.f, ss = 0.f;
    for (int k = 0; k < 16; ++k) { s += red1[tid * 16 + k]; ss += red2[tid * 16 + k]; }
    float mean = s / (float)D_MODEL;
    float var  = ss / (float)D_MODEL - mean * mean;
    stat_a[tid] = mean;
    stat_b[tid] = rsqrtf(var + 1e-5f);
  }
  __syncthreads();
  for (int i = tid; i < TOK_PER_BLK * D_MODEL; i += THREADS) {
    int m = i >> 9, c = i & (D_MODEL - 1);
    float v = (out_t[i] - stat_a[m]) * stat_b[m] * ln_w[c] + ln_b[c];
    out[(size_t)(t0 + m) * D_MODEL + c] = v;
  }
}

extern "C" void kernel_launch(void* const* d_in, const int* in_sizes, int n_in,
                              void* d_out, int out_size, void* d_ws, size_t ws_size,
                              hipStream_t stream) {
  (void)in_sizes; (void)n_in; (void)out_size; (void)ws_size;
  const float* x     = (const float*)d_in[0];
  const float* W_in  = (const float*)d_in[1];
  const float* b_in  = (const float*)d_in[2];
  const float* W_out = (const float*)d_in[3];
  const float* b_out = (const float*)d_in[4];
  const float* rot   = (const float*)d_in[5];
  const float* ent   = (const float*)d_in[6];
  const float* ln_w  = (const float*)d_in[7];
  const float* ln_b  = (const float*)d_in[8];
  float* out = (float*)d_out;

  _Float16* bf1 = (_Float16*)d_ws;                              // 1 MB
  _Float16* bf2 = bf1 + (size_t)NT1 * KT1 * 32 * 16;            // 1 MB

  prep_win <<<(NT1 * KT1 * 32 * 16) / 256, 256, 0, stream>>>(W_in,  bf1);
  prep_wout<<<(NT2 * KT2 * 32 * 16) / 256, 256, 0, stream>>>(W_out, bf2);

  hipFuncSetAttribute((const void*)qff_main,
                      hipFuncAttributeMaxDynamicSharedMemorySize, LDS_BYTES);
  qff_main<<<N_TOKENS / TOK_PER_BLK, THREADS, LDS_BYTES, stream>>>(
      x, bf1, bf2, b_in, b_out, rot, ent, ln_w, ln_b, out);
}
